// VQVAE_64063732187651
// MI455X (gfx1250) — matrix operand order repaired and checked by reading, not compiled, not run on hardware
//
#include <hip/hip_runtime.h>

// ---------------------------------------------------------------------------
// VQ-VAE quantization for MI455X (gfx1250, wave32, WMMA + TDM).
//   z        : [32, 64, 64, 64] f32   (B, C, H, W)
//   codebook : [1024, 64] f32         (K, C)
//   out      : z_q [32,64,64,64] f32 followed by scalar loss (8388609 floats)
// argmin_k ( ||e_k||^2 - 2 z.e_k ) via bf16 hi/lo split WMMA
// (hiA*hiB + hiA*loB + loA*hiB ~= fp32 dot). Codebook (hi|lo|e2, 260KB)
// staged into LDS by a single Tensor Data Mover descriptor per workgroup.
// Each wave owns 32 positions -> 12 WMMAs per A-tile load from LDS.
// Loss uses the identity min||z-e||^2 = best + ||z||^2, so z is read exactly
// once from HBM (67MB total traffic = bandwidth floor at 23.3 TB/s).
// ---------------------------------------------------------------------------

typedef __attribute__((ext_vector_type(16))) __bf16 v16bf;
typedef __attribute__((ext_vector_type(8)))  __bf16 v8bf;
typedef __attribute__((ext_vector_type(8)))  float  v8f;
typedef __attribute__((ext_vector_type(4)))  float  v4f;
typedef __attribute__((ext_vector_type(4)))  unsigned int v4u;
typedef __attribute__((ext_vector_type(4)))  int    v4i;
typedef __attribute__((ext_vector_type(8)))  int    v8i;

#define NC_    64
#define NK_    1024
#define TOTAL_ 8388608          // 32*64*4096
// workspace layout (bytes)
#define WS_HI    0              // bf16 [1024][64]  = 131072 B
#define WS_LO    131072         // bf16 [1024][64]  = 131072 B
#define WS_E2    262144         // f32  [1024]      =   4096 B
#define WS_PART  266240         // f32  [512] per-block loss partials
#define CB_BYTES 266240         // hi + lo + e2, contiguous
#define CB_EL8   (CB_BYTES / 8) // 33280 8-byte elements for the TDM tile

union V16u { v16bf v; v8bf h[2]; };

#if defined(__has_builtin)
#if __has_builtin(__builtin_amdgcn_tensor_load_to_lds)
#define USE_TDM 1
#endif
#endif

#define WMMA_BF16(A, Bv, Cv) \
  __builtin_amdgcn_wmma_f32_16x16x32_bf16(false, (A), false, (Bv), (short)0, \
                                          (Cv), false, false)

// --- pre-kernel: split codebook into bf16 hi/lo planes, compute ||e||^2 ----
__global__ __launch_bounds__(256) void vq_prep(const float* __restrict__ cb,
                                               char* __restrict__ ws) {
  __bf16* cb_hi = (__bf16*)(ws + WS_HI);
  __bf16* cb_lo = (__bf16*)(ws + WS_LO);
  float*  e2    = (float*)(ws + WS_E2);
  int k = blockIdx.x * blockDim.x + threadIdx.x;
  if (k < NK_) {
    float s = 0.0f;
    for (int c = 0; c < NC_; ++c) {
      float x = cb[k * NC_ + c];
      s = fmaf(x, x, s);
      __bf16 hv = (__bf16)x;
      cb_hi[k * NC_ + c] = hv;
      cb_lo[k * NC_ + c] = (__bf16)(x - (float)hv);
    }
    e2[k] = s;
  }
}

// --- main kernel: 512 blocks x 256 threads (8 waves), 256 positions/block --
__global__ __launch_bounds__(256) void vq_main(const float* __restrict__ z,
                                               const float* __restrict__ cb,
                                               const char* __restrict__ ws_cb,
                                               float* __restrict__ out,
                                               float* __restrict__ partial) {
  extern __shared__ char smem[];               // hi(128K) | lo(128K) | e2(4K)
  __shared__ float red[8];

#ifdef USE_TDM
  // ---- Tensor Data Mover: one 1-D descriptor stages the whole 260KB ------
  if (threadIdx.x < 32) {                      // wave 0 issues the DMA
    unsigned long long gaddr = (unsigned long long)(const void*)ws_cb;
    unsigned lds_base = (unsigned)(unsigned long long)(void*)smem;
    v4u g0;
    g0[0] = 1u;                                   // count=1, user descriptor
    g0[1] = lds_base;                             // lds_addr (bytes)
    g0[2] = (unsigned)(gaddr & 0xFFFFFFFFull);    // global_addr[31:0]
    g0[3] = (unsigned)((gaddr >> 32) & 0x1FFFFFFull) | (2u << 30); // type=2
    v8i g1 = {};
    g1[0] = 0x00030000;                           // data_size=3 (8B elements)
    g1[1] = (int)((CB_EL8 & 0xFFFF) << 16);       // tensor_dim0[15:0]
    g1[2] = (int)(((CB_EL8 >> 16) & 0xFFFF) | (1u << 16)); // tdim0 hi | tdim1=1
    g1[3] = (int)((CB_EL8 & 0xFFFF) << 16);       // tile_dim0 = tensor_dim0
    g1[4] = 0;                                    // tile_dim1/2 unused
    g1[5] = (int)CB_EL8;                          // tensor_dim0_stride
    g1[6] = 0; g1[7] = 0;
    v4i g2 = {}; v4i g3 = {};
#if __clang_major__ >= 23
    v8i g2b = {};
    __builtin_amdgcn_tensor_load_to_lds(g0, g1, g2, g3, g2b, 0);
#else
    __builtin_amdgcn_tensor_load_to_lds(g0, g1, g2, g3, 0);
#endif
    __builtin_amdgcn_s_wait_tensorcnt(0);
  }
#else
  {                                            // fallback: cooperative copy
    const uint4* g = (const uint4*)ws_cb;
    uint4*       s = (uint4*)smem;
    for (int i = threadIdx.x; i < CB_BYTES / 16; i += 256) s[i] = g[i];
  }
#endif
  __syncthreads();

  const int n0   = blockIdx.x << 8;            // first flat position
  const int b    = n0 >> 12;                   // image index
  const int hw0  = n0 & 4095;                  // flat h*W+w base
  const int w    = threadIdx.x >> 5;           // wave id (0..7)
  const int lane = threadIdx.x & 31;
  const int lh   = lane >> 4;                  // half-wave (0/1)
  const int ln   = lane & 15;                  // position within 16-tile

  // --- load this wave's 32 z-vectors as two WMMA B fragment sets ----------
  // B layout (32x16, bf16): lane l holds column N=l%16; halfword i holds
  // K = i + 16*(l>>4). k-step s covers channels 32s..32s+31.
  // Also accumulate ||z||^2 (this lane covers half the channels; lane l^16
  // holds the other half -> completed by the shfl_xor(16) merge below).
  const float* zp0 = z + ((size_t)b << 18) + (size_t)(hw0 + (w << 5) + ln);
  const float* zp1 = zp0 + 16;
  v16bf bhA[2], blA[2], bhB[2], blB[2];
  float z2A = 0.0f, z2B = 0.0f;
#pragma unroll
  for (int s = 0; s < 2; ++s) {
#pragma unroll
    for (int i = 0; i < 16; ++i) {
      int c = (s << 5) + (lh << 4) + i;
      float xa = zp0[(size_t)c << 12];         // stride HW=4096 floats
      float xb = zp1[(size_t)c << 12];
      z2A = fmaf(xa, xa, z2A);
      z2B = fmaf(xb, xb, z2B);
      __bf16 ha = (__bf16)xa, hb = (__bf16)xb;
      bhA[s][i] = ha;  blA[s][i] = (__bf16)(xa - (float)ha);
      bhB[s][i] = hb;  blB[s][i] = (__bf16)(xb - (float)hb);
    }
  }

  const __bf16* lds_hi = (const __bf16*)smem;
  const __bf16* lds_lo = (const __bf16*)(smem + WS_LO);
  const float*  lds_e2 = (const float*)(smem + WS_E2);

  float bestA = 3.402823e38f, bestB = 3.402823e38f;
  int   idxA = 0, idxB = 0;

  // --- sweep 64 code tiles of 16; A = codebook tile (M=codes) -------------
  // A layout (16x32, bf16): lane l holds row M=l%16; halfword i holds
  // K = i + 16*(l>>4) -> one contiguous 16-elem run (matches B).
  for (int t = 0; t < 64; ++t) {
    const int krow = (t << 4) + ln;
    const v8bf* rh = (const v8bf*)(lds_hi + krow * NC_);
    const v8bf* rl = (const v8bf*)(lds_lo + krow * NC_);
    V16u ah0, ah1, al0, al1;
    ah0.h[0] = rh[2 * lh + 0]; ah0.h[1] = rh[2 * lh + 1];   // k-step 0: c in [0,32)
    ah1.h[0] = rh[2 * lh + 4]; ah1.h[1] = rh[2 * lh + 5];   // k-step 1: c in [32,64)
    al0.h[0] = rl[2 * lh + 0]; al0.h[1] = rl[2 * lh + 1];
    al1.h[0] = rl[2 * lh + 4]; al1.h[1] = rl[2 * lh + 5];

    v4f e2a = *(const v4f*)(lds_e2 + (t << 4) + (lh << 3));
    v4f e2b = *(const v4f*)(lds_e2 + (t << 4) + (lh << 3) + 4);

    // 12 WMMAs per A-tile: two position groups x (hi*hi + hi*lo + lo*hi)
    v8f accA = {}, accB = {};
    accA = WMMA_BF16(ah0.v, bhA[0], accA);  accB = WMMA_BF16(ah0.v, bhB[0], accB);
    accA = WMMA_BF16(ah1.v, bhA[1], accA);  accB = WMMA_BF16(ah1.v, bhB[1], accB);
    accA = WMMA_BF16(ah0.v, blA[0], accA);  accB = WMMA_BF16(ah0.v, blB[0], accB);
    accA = WMMA_BF16(ah1.v, blA[1], accA);  accB = WMMA_BF16(ah1.v, blB[1], accB);
    accA = WMMA_BF16(al0.v, bhA[0], accA);  accB = WMMA_BF16(al0.v, bhB[0], accB);
    accA = WMMA_BF16(al1.v, bhA[1], accA);  accB = WMMA_BF16(al1.v, bhB[1], accB);

    // D layout: acc[v] = (code 16t + v + 8*lh, position l%16)
#pragma unroll
    for (int v = 0; v < 8; ++v) {
      float ev = (v < 4) ? e2a[v] : e2b[v - 4];
      int  idx = (t << 4) + (lh << 3) + v;
      float dA = ev - 2.0f * accA[v];
      float dB = ev - 2.0f * accB[v];
      if (dA < bestA) { bestA = dA; idxA = idx; }
      if (dB < bestB) { bestB = dB; idxB = idx; }
    }
  }

  // merge the two code-halves (and the two ||z||^2 halves) held by lanes
  // l and l^16 (same position, complementary channels)
  {
    float oA = __shfl_xor(bestA, 16);  int iA = __shfl_xor(idxA, 16);
    float oB = __shfl_xor(bestB, 16);  int iB = __shfl_xor(idxB, 16);
    if (oA < bestA || (oA == bestA && iA < idxA)) { bestA = oA; idxA = iA; }
    if (oB < bestB || (oB == bestB && iB < idxB)) { bestB = oB; idxB = iB; }
    z2A += __shfl_xor(z2A, 16);
    z2B += __shfl_xor(z2B, 16);
  }

  // loss per position: min_k ||z - e_k||^2 = best + ||z||^2
  // (count each position once: lanes 0..15 only)
  float lsum = (lh == 0) ? (bestA + z2A) + (bestB + z2B) : 0.0f;

  // --- gather z_q rows (L2-hot, b128-vectorizable) and write coalesced ----
  const size_t obase = ((size_t)b << 18) + (size_t)(hw0 + (w << 5) + ln);
  const float* qA = cb + (size_t)idxA * NC_;
  const float* qB = cb + (size_t)idxB * NC_;
#pragma unroll 4
  for (int it = 0; it < 32; ++it) {
    int c = it + (lh << 5);                    // contiguous per lane -> b128
    size_t o0 = obase + ((size_t)c << 12);
    out[o0]      = qA[c];
    out[o0 + 16] = qB[c];
  }

  // deterministic loss reduction: wave shfl tree -> LDS -> thread 0
#pragma unroll
  for (int off = 16; off > 0; off >>= 1) lsum += __shfl_xor(lsum, off);
  if (lane == 0) red[w] = lsum;
  __syncthreads();
  if (threadIdx.x == 0) {
    float s = 0.0f;
#pragma unroll
    for (int i = 0; i < 8; ++i) s += red[i];
    partial[blockIdx.x] = s;
  }
}

// --- final fixed-order loss reduce: 512 partials -> scalar -----------------
__global__ void vq_loss(const float* __restrict__ partial,
                        float* __restrict__ loss_out) {
  int l = threadIdx.x;                   // 32 threads, one wave
  float s = 0.0f;
  for (int i = 0; i < 16; ++i) s += partial[l + 32 * i];
#pragma unroll
  for (int off = 16; off > 0; off >>= 1) s += __shfl_xor(s, off);
  if (l == 0) *loss_out = s * (1.25f / 8388608.0f);   // (1+BETA)*mean
}

extern "C" void kernel_launch(void* const* d_in, const int* in_sizes, int n_in,
                              void* d_out, int out_size, void* d_ws, size_t ws_size,
                              hipStream_t stream) {
  (void)in_sizes; (void)n_in; (void)out_size; (void)ws_size;
  const float* z  = (const float*)d_in[0];
  const float* cb = (const float*)d_in[1];
  float* out = (float*)d_out;
  char*  ws  = (char*)d_ws;
  float* partial = (float*)(ws + WS_PART);

  vq_prep<<<4, 256, 0, stream>>>(cb, ws);
  vq_main<<<512, 256, CB_BYTES, stream>>>(z, cb, ws, out, partial);
  vq_loss<<<1, 32, 0, stream>>>(partial, out + TOTAL_);
}
